// KG_infuded_module_91096256348255
// MI455X (gfx1250) — compile-verified, matrix operand order checked by reading, not run on hardware
//
#include <hip/hip_runtime.h>
#include <math.h>

// ---- problem constants (match reference) ----
#define BQ 4
#define SEQ 2048
#define HD 4096
#define WW 512
#define EE 4
#define TT 4
#define KD 100
#define KDP 128
#define KI 1024
#define FEATK 4224   // HD + KDP, multiple of 64
#define MLPK 4196    // HD + KD

typedef __attribute__((ext_vector_type(16))) __bf16 v16bf;
typedef __attribute__((ext_vector_type(8)))  float  v8f;

union Frag {
  v16bf v;
  uint4 u[2];
};

__device__ __forceinline__ unsigned short f2bf(float f) {
  union { float f; unsigned u; } v; v.f = f;
  unsigned r = (v.u + 0x7FFFu + ((v.u >> 16) & 1u)) >> 16;
  return (unsigned short)r;
}
__device__ __forceinline__ float bf2f(unsigned short s) {
  union { unsigned u; float f; } v; v.u = ((unsigned)s) << 16;
  return v.f;
}
__device__ __forceinline__ float siluf(float x) {
  return x * (1.0f / (1.0f + expf(-x)));
}

// CDNA5 async global->LDS copy, 16B per lane, tracked by ASYNCcnt.
// VGPR[VDST] = workgroup-relative LDS byte address = low 32 bits of the
// generic shared pointer.
__device__ __forceinline__ void async_load16(const void* g, void* lds) {
  unsigned l = (unsigned)(uintptr_t)lds;
  asm volatile("global_load_async_to_lds_b128 %0, %1, off"
               :: "v"(l), "v"(g) : "memory");
}

// =====================================================================
// Generic bf16 WMMA GEMM:  C[M,N] = A[M,K] @ W[N,K]^T    (A row-major,
// W "torch Linear" layout). M%128==0, N%128==0, K%32==0 guaranteed by caller.
// Double-buffered LDS, async staging overlapped with WMMA compute.
// epilogue: 0 = store f32, 1 = store bf16,
//           2 = f32 silu(acc+bias[n])*alpha + residual[m,n]   (final MLP)
//           3 = bf16 silu(acc) * bf2f(Ubf[m,n])               (gate*up fuse)
// =====================================================================
#define BM 128
#define BN 128
#define BK 32
#define LDT 40   // padded LDS row stride (bf16 elems); 80 bytes, 16B aligned

__global__ __launch_bounds__(256) void gemm_bf16_wmma(
    const unsigned short* __restrict__ A, int lda,
    const unsigned short* __restrict__ Wm, int ldb,
    void* __restrict__ Cout, int M, int N, int K,
    int epilogue,
    const float* __restrict__ bias,
    const float* __restrict__ residual,
    const float* __restrict__ alphap,
    const unsigned short* __restrict__ Ubf)
{
  __shared__ unsigned short As[2][BM * LDT];
  __shared__ unsigned short Bs[2][BN * LDT];

  const int t    = threadIdx.x;
  const int lane = t & 31;
  const int wid  = t >> 5;
  const int wm   = wid & 1;        // 2 wave-rows  (64 rows each)
  const int wn   = wid >> 1;       // 4 wave-cols  (32 cols each)
  const int half = lane >> 4;
  const int l16  = lane & 15;
  const int m0   = blockIdx.y * BM;
  const int n0   = blockIdx.x * BN;

  // tile staging: 128 rows x 32 bf16 = 512 x 16B chunks per matrix;
  // 2 chunks per thread per matrix -> 4 async instructions per wave per stage
  auto stage = [&](int kt, int buf) {
    const int k0 = kt * BK;
#pragma unroll
    for (int i = 0; i < 2; ++i) {
      int ch  = t + 256 * i;
      int row = ch >> 2;
      int seg = ch & 3;
      async_load16(&A[(size_t)(m0 + row) * lda + k0 + seg * 8],
                   &As[buf][row * LDT + seg * 8]);
      async_load16(&Wm[(size_t)(n0 + row) * ldb + k0 + seg * 8],
                   &Bs[buf][row * LDT + seg * 8]);
    }
  };

  v8f acc[4][2];
  v8f zero = {};
#pragma unroll
  for (int mi = 0; mi < 4; ++mi)
#pragma unroll
    for (int ni = 0; ni < 2; ++ni) acc[mi][ni] = zero;

  const int ktiles = K / BK;
  stage(0, 0);
  for (int kt = 0; kt < ktiles; ++kt) {
    const int cur = kt & 1;
    if (kt + 1 < ktiles) {
      stage(kt + 1, cur ^ 1);
      // previous stage's 4 ops complete when <=4 remain (in-order completion)
      asm volatile("s_wait_asynccnt 0x4" ::: "memory");
    } else {
      asm volatile("s_wait_asynccnt 0x0" ::: "memory");
    }
    __syncthreads();

    Frag a[4], b[2];
#pragma unroll
    for (int mi = 0; mi < 4; ++mi) {
      int m = wm * 64 + mi * 16 + l16;
      int base = m * LDT + half * 8;           // K 0-7 / 8-15 by lane half
      a[mi].u[0] = *(const uint4*)&As[cur][base];
      a[mi].u[1] = *(const uint4*)&As[cur][base + 16];  // K 16-23 / 24-31
    }
#pragma unroll
    for (int ni = 0; ni < 2; ++ni) {
      int n = wn * 32 + ni * 16 + l16;
      int base = n * LDT + half * 8;
      b[ni].u[0] = *(const uint4*)&Bs[cur][base];
      b[ni].u[1] = *(const uint4*)&Bs[cur][base + 16];
    }
#pragma unroll
    for (int mi = 0; mi < 4; ++mi)
#pragma unroll
      for (int ni = 0; ni < 2; ++ni)
        acc[mi][ni] = __builtin_amdgcn_wmma_f32_16x16x32_bf16(
            false, a[mi].v, false, b[ni].v, (short)0, acc[mi][ni],
            false, false);
    __syncthreads();   // all waves done reading 'cur' before it is restaged
  }

  const size_t Ns = (size_t)N;
#pragma unroll
  for (int mi = 0; mi < 4; ++mi)
#pragma unroll
    for (int ni = 0; ni < 2; ++ni)
#pragma unroll
      for (int i = 0; i < 8; ++i) {
        int m = m0 + wm * 64 + mi * 16 + half * 8 + i;  // C/D layout: M=vgpr(+8)
        int n = n0 + wn * 32 + ni * 16 + l16;           // N = lane%16
        float v = acc[mi][ni][i];
        size_t off = (size_t)m * Ns + n;
        if (epilogue == 0) {
          ((float*)Cout)[off] = v;
        } else if (epilogue == 1) {
          ((unsigned short*)Cout)[off] = f2bf(v);
        } else if (epilogue == 2) {
          v += bias[n];
          ((float*)Cout)[off] = siluf(v) * alphap[0] + residual[off];
        } else {
          float uu = bf2f(Ubf[off]);
          ((unsigned short*)Cout)[off] = f2bf(siluf(v) * uu);
        }
      }
}

// ---------------------------------------------------------------------
// fp32 -> bf16 with column pad: dst[R,Cp], src[R,C]
__global__ void conv_pad_bf16(const float* __restrict__ src,
                              unsigned short* __restrict__ dst,
                              int R, int C, int Cp) {
  size_t total = (size_t)R * Cp;
  for (size_t idx = (size_t)blockIdx.x * blockDim.x + threadIdx.x; idx < total;
       idx += (size_t)gridDim.x * blockDim.x) {
    size_t r = idx / Cp, c = idx % Cp;
    dst[idx] = (c < (size_t)C) ? f2bf(src[r * C + c]) : (unsigned short)0;
  }
}

// transpose down_w [HD,KI] fp32 -> [KI,HD] bf16  (weight layout for A@W^T GEMM)
__global__ void transpose_down(const float* __restrict__ src,
                               unsigned short* __restrict__ dst) {
  size_t total = (size_t)HD * KI;
  for (size_t idx = (size_t)blockIdx.x * blockDim.x + threadIdx.x; idx < total;
       idx += (size_t)gridDim.x * blockDim.x) {
    size_t h = idx / KI, j = idx % KI;
    dst[j * HD + h] = f2bf(src[idx]);
  }
}

// RMSNorm row -> feat bf16 cols [0,HD); zero kg cols [HD,FEATK)
__global__ __launch_bounds__(256) void rmsnorm_kernel(
    const float* __restrict__ hidden, const float* __restrict__ w,
    unsigned short* __restrict__ feat) {
  __shared__ float red[256];
  const int row = blockIdx.x;          // 0..B*S-1
  const int t = threadIdx.x;
  const float* x = hidden + (size_t)row * HD;
  float p = 0.f;
#pragma unroll
  for (int j = 0; j < HD / 256; ++j) { float v = x[t + 256 * j]; p += v * v; }
  red[t] = p; __syncthreads();
  for (int off = 128; off > 0; off >>= 1) {
    if (t < off) red[t] += red[t + off];
    __syncthreads();
  }
  float scale = rsqrtf(red[0] / (float)HD + 1e-6f);
  unsigned short* frow = feat + (size_t)row * FEATK;
#pragma unroll
  for (int j = 0; j < HD / 256; ++j) {
    int c = t + 256 * j;
    frow[c] = f2bf(w[c] * x[c] * scale);
  }
  if (t < FEATK - HD) frow[HD + t] = 0;  // kg columns start zero
}

// gather entity embeddings -> ent_bf [B*W*5, KDP] bf16 (pad K 100->128)
__global__ void ent_gather(const int* __restrict__ words_ents,
                           const float* __restrict__ concept_embed,
                           const float* __restrict__ sentinel,
                           unsigned short* __restrict__ ent_bf) {
  const int r = blockIdx.x;            // (bq*W + w)*5 + e
  const int t = threadIdx.x;           // 0..127
  const int e = r % 5;
  const int bw = r / 5;                // bq*W + w
  float v = 0.f;
  if (t < KD) {
    if (e < EE) {
      int we = words_ents[(size_t)bw * EE + e];
      int idx = we < 0 ? 0 : we;
      v = concept_embed[(size_t)idx * KD + t];
    } else {
      v = sentinel[t];
    }
  }
  ent_bf[(size_t)r * KDP + t] = f2bf(v);
}

// word-mean of normalized hidden -> b_bf [B*W, HD] bf16
__global__ __launch_bounds__(256) void word_mean(
    const unsigned short* __restrict__ feat, const int* __restrict__ wst,
    unsigned short* __restrict__ b_bf) {
  const int bw = blockIdx.x;           // bq*W + w
  const int bq = bw / WW;
  const int t = threadIdx.x;
  int s[TT]; float cnt = 0.f;
#pragma unroll
  for (int k = 0; k < TT; ++k) {
    s[k] = wst[(size_t)bw * TT + k];
    if (s[k] >= 0) cnt += 1.f;
  }
  float inv = 1.f / cnt;
#pragma unroll
  for (int j = 0; j < HD / 256; ++j) {
    int c = t + 256 * j;
    float acc = 0.f;
#pragma unroll
    for (int k = 0; k < TT; ++k)
      if (s[k] >= 0)
        acc += bf2f(feat[((size_t)bq * SEQ + s[k]) * FEATK + c]);
    b_bf[(size_t)bw * HD + c] = f2bf(acc * inv);
  }
}

// attention: aw[e]=h(b,w,e)·c(b,w); masked softmax; attn_out scattered to feat kg cols
__global__ __launch_bounds__(256) void attn_scatter(
    const unsigned short* __restrict__ h_bf, const float* __restrict__ c,
    const int* __restrict__ words_ents, const int* __restrict__ wst,
    const float* __restrict__ concept_embed, const float* __restrict__ sentinel,
    unsigned short* __restrict__ feat) {
  __shared__ float red[256];
  __shared__ float attnS[5];
  const int bw = blockIdx.x;           // bq*W + w
  const int bq = bw / WW;
  const int t = threadIdx.x;
  const float* crow = c + (size_t)bw * KI;

  float part[5];
#pragma unroll
  for (int e = 0; e < 5; ++e) {
    const unsigned short* hrow = h_bf + ((size_t)bw * 5 + e) * KI;
    float p = 0.f;
#pragma unroll
    for (int j = 0; j < KI / 256; ++j) {
      int i = t + 256 * j;
      p += bf2f(hrow[i]) * crow[i];
    }
    part[e] = p;
  }
  float aw[5];
  for (int e = 0; e < 5; ++e) {
    red[t] = part[e]; __syncthreads();
    for (int off = 128; off > 0; off >>= 1) {
      if (t < off) red[t] += red[t + off];
      __syncthreads();
    }
    aw[e] = red[0];
    __syncthreads();
  }
  if (t == 0) {
#pragma unroll
    for (int e = 0; e < EE; ++e)
      if (words_ents[(size_t)bw * EE + e] == -1) aw[e] += -1e9f;
    float mx = aw[0];
    for (int e = 1; e < 5; ++e) mx = fmaxf(mx, aw[e]);
    float sum = 0.f;
    for (int e = 0; e < 5; ++e) { aw[e] = expf(aw[e] - mx); sum += aw[e]; }
    float inv = 1.f / sum;
    for (int e = 0; e < 5; ++e) attnS[e] = aw[e] * inv;
  }
  __syncthreads();

  if (t < KD) {
    float o = 0.f;
#pragma unroll
    for (int e = 0; e < 5; ++e) {
      float ev;
      if (e < EE) {
        int we = words_ents[(size_t)bw * EE + e];
        int idx = we < 0 ? 0 : we;
        ev = concept_embed[(size_t)idx * KD + t];
      } else {
        ev = sentinel[t];
      }
      o += attnS[e] * ev;
    }
    unsigned short ob = f2bf(o);
#pragma unroll
    for (int k = 0; k < TT; ++k) {
      int s = wst[(size_t)bw * TT + k];
      if (s >= 0)
        feat[((size_t)bq * SEQ + s) * FEATK + HD + t] = ob;
    }
  }
}

// =====================================================================
extern "C" void kernel_launch(void* const* d_in, const int* in_sizes, int n_in,
                              void* d_out, int out_size, void* d_ws, size_t ws_size,
                              hipStream_t stream) {
  const float* hidden   = (const float*)d_in[0];
  const float* cembed   = (const float*)d_in[1];
  const float* sentinel = (const float*)d_in[2];
  const float* rmsw     = (const float*)d_in[3];
  const float* gate_w   = (const float*)d_in[4];
  const float* up_w     = (const float*)d_in[5];
  const float* down_w   = (const float*)d_in[6];
  const float* mlp_w    = (const float*)d_in[7];
  const float* mlp_b    = (const float*)d_in[8];
  const float* alpha    = (const float*)d_in[9];
  const int*   wents    = (const int*)d_in[10];
  const int*   wst      = (const int*)d_in[11];
  float* out = (float*)d_out;

  char* ws = (char*)d_ws;
  // workspace layout (all offsets 256B aligned)
  unsigned short* feat_bf  = (unsigned short*)(ws + 0);                 // [8192,4224]
  unsigned short* mlpw_bf  = (unsigned short*)(ws + 69206016);          // [4096,4224]
  unsigned short* b_bf     = (unsigned short*)(ws + 103809024);         // [2048,4096]
  unsigned short* downT_bf = (unsigned short*)(ws + 120586240);         // [1024,4096]
  float*          c_f32    = (float*)(ws + 128974848);                  // [2048,1024]
  unsigned short* gate_bf  = (unsigned short*)(ws + 137363456);         // [1024,128]
  unsigned short* up_bf    = (unsigned short*)(ws + 137625600);         // [1024,128]
  unsigned short* ent_bf   = (unsigned short*)(ws + 137887744);         // [10240,128]
  unsigned short* u_bf     = (unsigned short*)(ws + 140509184);         // [10240,1024]
  unsigned short* h_bf     = (unsigned short*)(ws + 161480704);         // [10240,1024]

  const int MROWS = BQ * SEQ;   // 8192
  const int NWORD = BQ * WW;    // 2048
  const int NENT  = NWORD * 5;  // 10240

  // weight prep
  conv_pad_bf16<<<512, 256, 0, stream>>>(gate_w, gate_bf, KI, KD, KDP);
  conv_pad_bf16<<<512, 256, 0, stream>>>(up_w, up_bf, KI, KD, KDP);
  conv_pad_bf16<<<8192, 256, 0, stream>>>(mlp_w, mlpw_bf, HD, MLPK, FEATK);
  transpose_down<<<8192, 256, 0, stream>>>(down_w, downT_bf);

  // activations prep
  rmsnorm_kernel<<<MROWS, 256, 0, stream>>>(hidden, rmsw, feat_bf);
  ent_gather<<<NENT, 128, 0, stream>>>(wents, cembed, sentinel, ent_bf);
  word_mean<<<NWORD, 256, 0, stream>>>(feat_bf, wst, b_bf);

  // c = b @ down_w : [2048,1024], K=4096
  {
    dim3 g(KI / BN, NWORD / BM);
    gemm_bf16_wmma<<<g, 256, 0, stream>>>(b_bf, HD, downT_bf, HD, c_f32,
                                          NWORD, KI, HD, 0,
                                          nullptr, nullptr, nullptr, nullptr);
  }
  // u = ent @ up^T (bf16); h = silu(ent @ gate^T) * u (bf16)   M=10240, K=128
  {
    dim3 g(KI / BN, NENT / BM);
    gemm_bf16_wmma<<<g, 256, 0, stream>>>(ent_bf, KDP, up_bf, KDP, u_bf,
                                          NENT, KI, KDP, 1,
                                          nullptr, nullptr, nullptr, nullptr);
    gemm_bf16_wmma<<<g, 256, 0, stream>>>(ent_bf, KDP, gate_bf, KDP, h_bf,
                                          NENT, KI, KDP, 3,
                                          nullptr, nullptr, nullptr, u_bf);
  }
  // attention + scatter into feat kg columns
  attn_scatter<<<NWORD, 256, 0, stream>>>(h_bf, c_f32, wents, wst, cembed,
                                          sentinel, feat_bf);
  // final: out = silu(feat @ mlp_w^T + b) * alpha + residual   M=8192,N=4096,K=4224
  {
    dim3 g(HD / BN, MROWS / BM);
    gemm_bf16_wmma<<<g, 256, 0, stream>>>(feat_bf, FEATK, mlpw_bf, FEATK, out,
                                          MROWS, HD, FEATK, 2,
                                          mlp_b, hidden, alpha, nullptr);
  }
}
